// SmolLM2Attention_6545530159358
// MI455X (gfx1250) — matrix operand order repaired and checked by reading, not compiled, not run on hardware
//
#include <hip/hip_runtime.h>
#include <hip/hip_bf16.h>

// MI455X / gfx1250: wave32, WMMA 16x16x32 bf16 -> f32 accumulate.
// All matmul operands pre-converted to bf16 once; inner loops are b128 loads + v_wmma.

typedef __attribute__((ext_vector_type(16))) __bf16 v16bf;
typedef __attribute__((ext_vector_type(8)))  __bf16 v8bf;
typedef __attribute__((ext_vector_type(8)))  float  v8f;

#define B_   4
#define S_   2048
#define HID_ 576
#define NH_  9
#define NKV_ 3
#define HD_  64

__device__ __forceinline__ __bf16 tobf(float x) { return (__bf16)x; }

// Build a 16-element bf16 fragment from two 16-byte chunks (lowers to b128 loads).
__device__ __forceinline__ v16bf load_frag16(const __bf16* p0, const __bf16* p1) {
  v8bf lo = *(const v8bf*)p0;
  v8bf hi = *(const v8bf*)p1;
  v16bf r;
#pragma unroll
  for (int e = 0; e < 8; ++e) { r[e] = lo[e]; r[e + 8] = hi[e]; }
  return r;
}

// ---------------------------------------------------------------------------
// C[M,N] = A[M,K] @ W[N,K]^T, A/W bf16 row-major, C f32. One wave per 16x16 tile.
// CDNA5 ISA 7.12.2 fragment layouts:
//   A 16x32: lane -> row (L&15); e<8 -> k=kA0+e, e>=8 -> k=kA0+16+(e-8); kA0=(L>=16)?8:0
//   B 32x16: lane -> col (L&15); e<8 -> k=kB0+e, e>=8 -> k=kB0+16+(e-8); kB0=(L>=16)?8:0
//   C/D:     lane -> col (L&15); vgpr r -> row r + 8*(L>=16)
// ---------------------------------------------------------------------------
__global__ __launch_bounds__(128)
void wmma_gemm_bf16(const __bf16* __restrict__ A, const __bf16* __restrict__ W,
                    float* __restrict__ C, int M, int N, int K) {
  const int lane = threadIdx.x & 31;
  const int wave = blockIdx.x * (blockDim.x >> 5) + (threadIdx.x >> 5);
  const int tilesN = N >> 4;
  if (wave >= (M >> 4) * tilesN) return;   // wave-uniform: EXEC stays all-ones
  const int tm = wave / tilesN, tn = wave % tilesN;
  const int l15 = lane & 15;
  const int hi  = lane >> 4;
  const int kA0 = hi * 8;
  const int kB0 = hi * 8;

  const __bf16* ap = A + (size_t)(tm * 16 + l15) * K;
  const __bf16* bp = W + (size_t)(tn * 16 + l15) * K;

  v8f acc = {};
  for (int k0 = 0; k0 < K; k0 += 32) {
    __builtin_prefetch(ap + k0 + 128, 0, 1);   // global_prefetch_b8
    v16bf a = load_frag16(ap + k0 + kA0, ap + k0 + kA0 + 16);
    v16bf b = load_frag16(bp + k0 + kB0, bp + k0 + kB0 + 16);
    acc = __builtin_amdgcn_wmma_f32_16x16x32_bf16(false, a, false, b,
                                                  (short)0, acc, false, false);
  }
  const int mBase = tm * 16 + hi * 8;
  const int n = tn * 16 + l15;
#pragma unroll
  for (int r = 0; r < 8; ++r) C[(size_t)(mBase + r) * N + n] = acc[r];
}

// ---------------------------------------------------------------------------
// Elementwise f32 -> bf16 with scale (scale=1 for plain conversion).
// ---------------------------------------------------------------------------
__global__ void cvt_bf16_kernel(const float* __restrict__ src, __bf16* __restrict__ dst,
                                int n, float scale) {
  int i = blockIdx.x * blockDim.x + threadIdx.x;
  if (i < n) dst[i] = tobf(src[i] * scale);
}

// ---------------------------------------------------------------------------
// RoPE (rotate-half), in place on f32. X: [B*S][heads*64]. fc: [S][32][2] (cos,sin)
// ---------------------------------------------------------------------------
__global__ void rope_kernel(float* __restrict__ X, const float* __restrict__ fc, int heads) {
  int idx = blockIdx.x * blockDim.x + threadIdx.x;
  int total = B_ * S_ * heads * (HD_ / 2);
  if (idx >= total) return;
  int d   = idx & 31;
  int hh  = (idx >> 5) % heads;
  int row = idx / (32 * heads);           // b*S + s
  int s   = row % S_;
  float c  = fc[((size_t)s * 32 + d) * 2 + 0];
  float sn = fc[((size_t)s * 32 + d) * 2 + 1];
  float* p = X + (size_t)row * heads * HD_ + hh * HD_;
  float x1 = p[d], x2 = p[d + 32];
  p[d]      = x1 * c - x2 * sn;
  p[d + 32] = x2 * c + x1 * sn;
}

// ---------------------------------------------------------------------------
// Vt_bf16[(b*NKV+kh)*64 + d][s] = V_f32[b*S+s][kh*64+d]
// ---------------------------------------------------------------------------
__global__ void transpose_v_bf16(const float* __restrict__ V, __bf16* __restrict__ Vt) {
  int idx = blockIdx.x * blockDim.x + threadIdx.x;
  const int total = B_ * S_ * NKV_ * HD_;
  if (idx >= total) return;
  int d   = idx % HD_;
  int kh  = (idx / HD_) % NKV_;
  int row = idx / (NKV_ * HD_);           // b*S + s
  int bq  = row / S_;
  int s   = row % S_;
  Vt[((size_t)(bq * NKV_ + kh) * HD_ + d) * S_ + s] = tobf(V[idx]);
}

// ---------------------------------------------------------------------------
// Flash attention: one wave per (b, head, 16-query tile); full softmax over S=2048
// in 32-column chunks. All operands bf16 (Q pre-scaled by 0.125), accum f32.
// Output written as bf16 for the final wo GEMM.
// ---------------------------------------------------------------------------
#define FA_WAVES 4
__global__ __launch_bounds__(FA_WAVES * 32)
void flash_attn(const __bf16* __restrict__ Q, const __bf16* __restrict__ Kb,
                const __bf16* __restrict__ Vt, __bf16* __restrict__ O) {
  __shared__ float  sc[FA_WAVES][16][33];   // score chunk, padded rows
  __shared__ __bf16 pt[FA_WAVES][16][40];   // probs bf16, 80B row stride (16B aligned)
  __shared__ float  s_alpha[FA_WAVES][16];
  __shared__ float  s_l[FA_WAVES][16];

  const int lane = threadIdx.x & 31;
  const int wv   = threadIdx.x >> 5;
  const int task = blockIdx.x * FA_WAVES + wv;
  if (task >= B_ * NH_ * (S_ / 16)) return;   // wave-uniform
  const int qt = task % (S_ / 16);
  const int h  = (task / (S_ / 16)) % NH_;
  const int b  = task / ((S_ / 16) * NH_);
  const int kh = h / (NH_ / NKV_);

  const int l15   = lane & 15;
  const int hi    = lane >> 4;
  const int kA0   = hi * 8;
  const int kB0   = hi * 8;
  const int rbase = hi * 8;

  // Q fragments (already scaled by 1/sqrt(64))
  const __bf16* qp = Q + (size_t)(b * S_ + qt * 16 + l15) * (NH_ * HD_) + h * HD_;
  v16bf aq[2];
#pragma unroll
  for (int half = 0; half < 2; ++half)
    aq[half] = load_frag16(qp + half * 32 + kA0, qp + half * 32 + kA0 + 16);

  v8f oacc[4] = {};                 // 16 q-rows x 64 d-cols
  float m_i = -3.0e38f, l_i = 0.0f; // lanes 0..15 own row = lane

  const __bf16* kbase = Kb + (size_t)(b * S_) * (NKV_ * HD_) + kh * HD_;
  const __bf16* vbase = Vt + (size_t)(b * NKV_ + kh) * HD_ * S_;

  for (int sb = 0; sb < S_; sb += 32) {
    // ---- S = Q K^T for 32 key columns
    v8f s0 = {}, s1 = {};
#pragma unroll
    for (int dh = 0; dh < 2; ++dh) {
      const __bf16* kp = kbase + (size_t)(sb + l15) * (NKV_ * HD_) + dh * 32 + kB0;
      v16bf bk = load_frag16(kp, kp + 16);
      s0 = __builtin_amdgcn_wmma_f32_16x16x32_bf16(false, aq[dh], false, bk,
                                                   (short)0, s0, false, false);
      const __bf16* kp1 = kp + (size_t)16 * (NKV_ * HD_);
      v16bf bk1 = load_frag16(kp1, kp1 + 16);
      s1 = __builtin_amdgcn_wmma_f32_16x16x32_bf16(false, aq[dh], false, bk1,
                                                   (short)0, s1, false, false);
    }

    // ---- dump scores to LDS (same-wave ds ops are in-order; no barrier needed)
#pragma unroll
    for (int r = 0; r < 8; ++r) {
      sc[wv][rbase + r][l15]      = s0[r];
      sc[wv][rbase + r][16 + l15] = s1[r];
    }

    // ---- online softmax: lane < 16 owns row = lane
    if (lane < 16) {
      float mx = m_i;
#pragma unroll
      for (int j = 0; j < 32; ++j) mx = fmaxf(mx, sc[wv][lane][j]);
      float alpha = __expf(m_i - mx);
      float sum = 0.0f;
#pragma unroll
      for (int j = 0; j < 32; ++j) {
        float p = __expf(sc[wv][lane][j] - mx);
        pt[wv][lane][j] = tobf(p);
        sum += p;
      }
      l_i = l_i * alpha + sum;
      m_i = mx;
      s_alpha[wv][lane] = alpha;
    }

    // ---- rescale O accumulators
#pragma unroll
    for (int r = 0; r < 8; ++r) {
      float al = s_alpha[wv][rbase + r];
      oacc[0][r] *= al; oacc[1][r] *= al; oacc[2][r] *= al; oacc[3][r] *= al;
    }

    // ---- P as bf16 A-fragment straight from LDS (ds_load b128)
    const __bf16* pp = &pt[wv][l15][0];
    v16bf apf = load_frag16(pp + kA0, pp + kA0 + 16);

    // ---- O += P V
#pragma unroll
    for (int t = 0; t < 4; ++t) {
      const __bf16* vp = vbase + (size_t)(t * 16 + l15) * S_ + sb + kB0;
      v16bf bv = load_frag16(vp, vp + 16);
      oacc[t] = __builtin_amdgcn_wmma_f32_16x16x32_bf16(false, apf, false, bv,
                                                        (short)0, oacc[t], false, false);
    }
  }

  // ---- finalize: divide by l, store bf16 for the wo GEMM
  if (lane < 16) s_l[wv][lane] = l_i;
  __bf16* op = O + (size_t)(b * S_ + qt * 16) * (NH_ * HD_) + h * HD_;
#pragma unroll
  for (int r = 0; r < 8; ++r) {
    float inv = 1.0f / s_l[wv][rbase + r];
#pragma unroll
    for (int t = 0; t < 4; ++t)
      op[(size_t)(rbase + r) * (NH_ * HD_) + t * 16 + l15] = tobf(oacc[t][r] * inv);
  }
}

// ---------------------------------------------------------------------------
extern "C" void kernel_launch(void* const* d_in, const int* in_sizes, int n_in,
                              void* d_out, int out_size, void* d_ws, size_t ws_size,
                              hipStream_t stream) {
  const float* hs = (const float*)d_in[0];   // [4,2048,576]
  const float* fc = (const float*)d_in[1];   // [2048,32,2]
  const float* wq = (const float*)d_in[2];   // [576,576]
  const float* wk = (const float*)d_in[3];   // [192,576]
  const float* wv = (const float*)d_in[4];   // [192,576]
  const float* wo = (const float*)d_in[5];   // [576,576]
  float* out = (float*)d_out;                // [4,2048,576]

  const size_t R = (size_t)B_ * S_;          // 8192 rows

  // --- f32 scratch (GEMM outputs, RoPE in f32 for accuracy)
  float* wsf = (float*)d_ws;
  float* Qf = wsf;                           // R*576
  float* Kf = Qf + R * (NH_ * HD_);          // R*192
  float* Vf = Kf + R * (NKV_ * HD_);         // R*192
  float* f32_end = Vf + R * (NKV_ * HD_);

  // --- bf16 scratch (all WMMA operands)
  __bf16* wsb = (__bf16*)f32_end;
  __bf16* Hb  = wsb;                         // R*576
  __bf16* Wqb = Hb  + R * HID_;              // 576*576
  __bf16* Wkb = Wqb + (size_t)HID_ * HID_;   // 192*576
  __bf16* Wvb = Wkb + (size_t)(NKV_ * HD_) * HID_;
  __bf16* Wob = Wvb + (size_t)(NKV_ * HD_) * HID_;   // 576*576
  __bf16* Qb  = Wob + (size_t)HID_ * HID_;   // R*576 (pre-scaled)
  __bf16* Kb  = Qb  + R * (NH_ * HD_);       // R*192
  __bf16* Vtb = Kb  + R * (NKV_ * HD_);      // R*192 transposed
  __bf16* AOb = Vtb + R * (NKV_ * HD_);      // R*576

  const int M = (int)R;
  const int T = 256;

  // One-time bf16 conversion of activations + weights
  {
    int n = M * HID_;
    cvt_bf16_kernel<<<(n + T - 1) / T, T, 0, stream>>>(hs, Hb, n, 1.0f);
    n = HID_ * HID_;
    cvt_bf16_kernel<<<(n + T - 1) / T, T, 0, stream>>>(wq, Wqb, n, 1.0f);
    cvt_bf16_kernel<<<(n + T - 1) / T, T, 0, stream>>>(wo, Wob, n, 1.0f);
    n = (NKV_ * HD_) * HID_;
    cvt_bf16_kernel<<<(n + T - 1) / T, T, 0, stream>>>(wk, Wkb, n, 1.0f);
    cvt_bf16_kernel<<<(n + T - 1) / T, T, 0, stream>>>(wv, Wvb, n, 1.0f);
  }
  // QKV projections (bf16 operands, f32 out)
  {
    int waves = (M / 16) * (HID_ / 16);
    wmma_gemm_bf16<<<(waves + 3) / 4, 128, 0, stream>>>(Hb, Wqb, Qf, M, NH_ * HD_, HID_);
    waves = (M / 16) * ((NKV_ * HD_) / 16);
    wmma_gemm_bf16<<<(waves + 3) / 4, 128, 0, stream>>>(Hb, Wkb, Kf, M, NKV_ * HD_, HID_);
    wmma_gemm_bf16<<<(waves + 3) / 4, 128, 0, stream>>>(Hb, Wvb, Vf, M, NKV_ * HD_, HID_);
  }
  // RoPE in f32
  {
    int tq = M * NH_ * (HD_ / 2);
    rope_kernel<<<(tq + T - 1) / T, T, 0, stream>>>(Qf, fc, NH_);
    int tk = M * NKV_ * (HD_ / 2);
    rope_kernel<<<(tk + T - 1) / T, T, 0, stream>>>(Kf, fc, NKV_);
  }
  // Pack Q (scale 1/8 folded in), K, and transposed V to bf16
  {
    int n = M * NH_ * HD_;
    cvt_bf16_kernel<<<(n + T - 1) / T, T, 0, stream>>>(Qf, Qb, n, 0.125f);
    n = M * NKV_ * HD_;
    cvt_bf16_kernel<<<(n + T - 1) / T, T, 0, stream>>>(Kf, Kb, n, 1.0f);
    transpose_v_bf16<<<(n + T - 1) / T, T, 0, stream>>>(Vf, Vtb);
  }
  // Flash attention
  {
    int tasks = B_ * NH_ * (S_ / 16);
    flash_attn<<<(tasks + FA_WAVES - 1) / FA_WAVES, FA_WAVES * 32, 0, stream>>>(Qb, Kb, Vtb, AOb);
  }
  // Output projection into d_out
  {
    int waves = (M / 16) * (HID_ / 16);
    wmma_gemm_bf16<<<(waves + 3) / 4, 128, 0, stream>>>(AOb, Wob, out, M, HID_, HID_);
  }
}
